// ImageGS_90701119357568
// MI455X (gfx1250) — matrix operand order repaired and checked by reading, not compiled, not run on hardware
//
#include <hip/hip_runtime.h>

#define NG   1000          // real gaussians
#define NGP  1024          // padded (multiple of 32; pad entries are inert)
#define HH   512
#define WW   512
#define HWP  (HH * WW)

typedef __attribute__((ext_vector_type(16))) _Float16 v16h;
typedef __attribute__((ext_vector_type(8)))  float    v8f;

__device__ __forceinline__ float fast_exp2(float x) {
#if __has_builtin(__builtin_amdgcn_exp2f)
    return __builtin_amdgcn_exp2f(x);   // v_exp_f32
#else
    return exp2f(x);
#endif
}

__launch_bounds__(256)
__global__ void gs_render_wmma(const float* __restrict__ pos,   // [NG,2]
                               const float* __restrict__ scl,   // [NG,2]
                               const float* __restrict__ rot,   // [NG]
                               const float* __restrict__ col,   // [NG,3]
                               const float* __restrict__ opa,   // [NG]
                               float* __restrict__ out)         // [3,H,W]
{
    // Per-gaussian params, 32B/row: px, py, cos, sin, isx', isy', lop, 0
    //   isx' = sqrt(0.5*log2 e)/(|sx|+eps)  (folds exp() scaling)
    //   lop  = log2(sigmoid(op))            (folds opacity multiply)
    // alpha = exp2(lop - (xr*isx')^2 - (yr*isy')^2)
    __shared__ __align__(16) float    sG[NGP][8];        // 32768 B
    __shared__ __align__(4)  _Float16 sColB[4 * NGP];    //  8192 B, plane 3 == 0

    const int tid = threadIdx.x;

    // ---- cooperative precompute; pad entries evaluate to alpha == 0 ----
    for (int k = tid; k < NGP; k += 256) {
        float px = 0.f, py = 0.f, c = 0.f, s = 0.f;
        float isx = 0.f, isy = 0.f, lop = -1e30f;
        _Float16 c0 = (_Float16)0.f, c1 = (_Float16)0.f, c2 = (_Float16)0.f;
        if (k < NG) {
            px = pos[2 * k]; py = pos[2 * k + 1];
            const float kf = 0.84932180f;    // sqrt(0.5 * log2(e))
            isx = kf / (fabsf(scl[2 * k])     + 1e-6f);
            isy = kf / (fabsf(scl[2 * k + 1]) + 1e-6f);
            float r = rot[k];
            c = __cosf(r); s = __sinf(r);
            lop = -__log2f(1.0f + __expf(-opa[k]));      // log2(sigmoid(op))
            c0 = (_Float16)(1.0f / (1.0f + __expf(-col[3 * k + 0])));
            c1 = (_Float16)(1.0f / (1.0f + __expf(-col[3 * k + 1])));
            c2 = (_Float16)(1.0f / (1.0f + __expf(-col[3 * k + 2])));
        }
        sG[k][0] = px;  sG[k][1] = py;
        sG[k][2] = c;   sG[k][3] = s;
        sG[k][4] = isx; sG[k][5] = isy;
        sG[k][6] = lop; sG[k][7] = 0.0f;
        sColB[0 * NGP + k] = c0;
        sColB[1 * NGP + k] = c1;
        sColB[2 * NGP + k] = c2;
        sColB[3 * NGP + k] = (_Float16)0.f;   // zero plane for N = 3..15
    }
    __syncthreads();

    const int lane = tid & 31;
    const int wave = tid >> 5;
    const int hi   = lane >> 4;    // WMMA half-wave select
    const int mRow = lane & 15;    // A row M = pixel-in-strip
    const int nCol = lane & 15;    // B/D column N = channel

    const int base  = blockIdx.x * 128 + wave * 16;
    const int myPix = base + mRow;
    const float inv511 = 1.0f / 511.0f;
    const float fx = (float)(myPix & (WW - 1)) * inv511;
    const float fy = (float)(myPix >> 9)       * inv511;

    // Column pointer: channels 0-2 real, 3-15 -> zero plane (no cndmask in loop)
    const _Float16* __restrict__ cplane = &sColB[(nCol < 3 ? nCol : 3) * NGP];

    v8f acc = {0.f, 0.f, 0.f, 0.f, 0.f, 0.f, 0.f, 0.f};

#pragma unroll 1
    for (int g0 = 0; g0 < NGP; g0 += 32) {
        // Per-chunk lane base rows; all inner DS addrs are base + imm offset.
        const float* __restrict__ gbase = &sG[g0 + (hi << 3)][0];

        // ---- A tile: alpha[M][K] 16x32 f16 ----
        v16h A;
#pragma unroll
        for (int j = 0; j < 8; ++j) {
            // within-half K offsets: j=0..3 -> 0..7 ; j=4..7 -> 16..23
            const int offK = ((j >> 2) << 4) + ((j & 3) << 1);
#pragma unroll
            for (int e = 0; e < 2; ++e) {
                const float* __restrict__ gp = gbase + (offK + e) * 8;
                const float4 p0 = *(const float4*)(gp);
                const float4 p1 = *(const float4*)(gp + 4);
                const float dx = fx - p0.x;
                const float dy = fy - p0.y;
                const float xr = p0.z * dx + p0.w * dy;
                const float yr = p0.z * dy - p0.w * dx;
                const float tx = xr * p1.x;
                const float ty = yr * p1.y;
                float t = __builtin_fmaf(-tx, tx, p1.z);   // lop - tx^2
                t       = __builtin_fmaf(-ty, ty, t);      //     - ty^2
                A[2 * j + e] = (_Float16)fast_exp2(t);
            }
        }

        // ---- B tile: sigmoid(color)[K][N] 32x16 f16, pairs contiguous ----
        union { v16h h; unsigned int u[8]; } B;
        const _Float16* __restrict__ cbase = cplane + g0 + (hi << 3);
#pragma unroll
        for (int j = 0; j < 8; ++j)
            B.u[j] = *(const unsigned int*)(cbase + ((j >> 2) << 4) + ((j & 3) << 1));

        // D = A x B + C (f32 accum); TRANS-class -> co-executes with VALU.
        acc = __builtin_amdgcn_wmma_f32_16x16x32_f16(
            false, A, false, B.h, (short)0, acc, false, false);
    }

    // ---- D: VGPR r -> M = r + 8*hi, N = nCol; 8 consecutive pixels ----
    if (nCol < 3) {
        float* __restrict__ o = out + nCol * HWP + base + hi * 8;
        float4 o0, o1;
        o0.x = fminf(fmaxf(acc[0], 0.f), 1.f);
        o0.y = fminf(fmaxf(acc[1], 0.f), 1.f);
        o0.z = fminf(fmaxf(acc[2], 0.f), 1.f);
        o0.w = fminf(fmaxf(acc[3], 0.f), 1.f);
        o1.x = fminf(fmaxf(acc[4], 0.f), 1.f);
        o1.y = fminf(fmaxf(acc[5], 0.f), 1.f);
        o1.z = fminf(fmaxf(acc[6], 0.f), 1.f);
        o1.w = fminf(fmaxf(acc[7], 0.f), 1.f);
        *(float4*)(o)     = o0;
        *(float4*)(o + 4) = o1;
    }
}

extern "C" void kernel_launch(void* const* d_in, const int* in_sizes, int n_in,
                              void* d_out, int out_size, void* d_ws, size_t ws_size,
                              hipStream_t stream) {
    (void)in_sizes; (void)n_in; (void)out_size; (void)d_ws; (void)ws_size;
    const float* pos = (const float*)d_in[0];
    const float* scl = (const float*)d_in[1];
    const float* rot = (const float*)d_in[2];
    const float* col = (const float*)d_in[3];
    const float* opa = (const float*)d_in[4];
    float* out = (float*)d_out;

    dim3 grid(HWP / 128);   // 2048 blocks; 8 waves x 16-pixel strips
    dim3 block(256);
    gs_render_wmma<<<grid, block, 0, stream>>>(pos, scl, rot, col, opa, out);
}
